// BiMambaBlock_12000138625502
// MI455X (gfx1250) — compile-verified, matrix operand order checked by reading, not run on hardware
//
#include <hip/hip_runtime.h>
#include <hip/hip_bf16.h>

// ---------------------------------------------------------------------------
// BiMamba2 block for MI455X (gfx1250, wave32, WMMA).
// Dense contractions -> v_wmma_f32_16x16x32_bf16.
// GEMM tiles staged via global_load_async_to_lds_b128 (ASYNCcnt path),
// double-buffered in LDS.
// ---------------------------------------------------------------------------

typedef unsigned short u16;
typedef __bf16 v16bf __attribute__((ext_vector_type(16)));
typedef float  v8f   __attribute__((ext_vector_type(8)));

#define BATCH   2
#define SEQ     2048
#define DMODEL  1024
#define DINNER  2048
#define NHEADS  32
#define HEADDIM 64
#define DSTATE  128
#define NPROJ   4384      // 2*DINNER + 2*DSTATE + NHEADS
#define CONVCH  2304      // DINNER + 2*DSTATE
#define CHUNK   128
#define NCHUNK  16        // SEQ / CHUNK
#define NTOK    (BATCH*SEQ)   // 4096

// ---------------------- scalar helpers -------------------------------------
__device__ __forceinline__ u16 f2bf(float f) {
    union { float f; unsigned u; } v; v.f = f;
    unsigned u = v.u + (0x7FFFu + ((v.u >> 16) & 1u));   // round-nearest-even
    return (u16)(u >> 16);
}
__device__ __forceinline__ float bf2f(u16 h) {
    union { unsigned u; float f; } v; v.u = ((unsigned)h) << 16;
    return v.f;
}
__device__ __forceinline__ float siluf(float x) {
    return x / (1.f + __expf(-x));
}
__device__ __forceinline__ v8f v8f_zero() {
    v8f z = {0.f,0.f,0.f,0.f,0.f,0.f,0.f,0.f};
    return z;
}

// ---------------------- CDNA5 async global->LDS copy ------------------------
// 16B per lane, direct to LDS, tracked by ASYNCcnt (no VGPR round trip).
__device__ __forceinline__ void async_g2l_b128(void* ldsPtr, const void* gptr) {
    unsigned ldsOff = (unsigned)(size_t)ldsPtr;   // low 32b of shared aperture
    asm volatile("global_load_async_to_lds_b128 %0, %1, off"
                 :: "v"(ldsOff), "v"(gptr) : "memory");
}
__device__ __forceinline__ void wait_async0() {
    asm volatile("s_wait_asynccnt 0" ::: "memory");
}

// ---------------------- WMMA helpers ---------------------------------------
// 16x16x32 bf16 -> f32.  A fragment: [M=16][K=32] tile, row-major in LDS.
// B fragment: [N=16][K=32] tile (second operand pre-transposed so K is
// contiguous), same lane layout as A per ISA 7.12.2.
// Requires (stride*2) % 16 == 0 and (k0*2) % 16 == 0 for b128 LDS reads.
__device__ __forceinline__ v16bf ld_frag(const u16* __restrict__ tile,
                                         int row0, int k0, int stride) {
    int lane = threadIdx.x & 31;
    const u16* p = tile + (size_t)(row0 + (lane & 15)) * stride
                        + k0 + ((lane >> 4) << 3);
    union { v16bf v; uint4 q[2]; } f;
    f.q[0] = *(const uint4*)(p);        // K = kh .. kh+7
    f.q[1] = *(const uint4*)(p + 16);   // K = 16+kh .. 16+kh+7
    return f.v;
}
__device__ __forceinline__ v8f wmma_bf16(v16bf a, v16bf b, v8f c) {
    return __builtin_amdgcn_wmma_f32_16x16x32_bf16(
        false, a, false, b, (short)0, c, false, false);
}

// ---------------------- fp32 -> bf16 convert --------------------------------
__global__ void k_f32_to_bf16(const float* __restrict__ src,
                              u16* __restrict__ dst, int n) {
    int i = blockIdx.x * 256 + threadIdx.x;
    if (i < n) dst[i] = f2bf(src[i]);
}

// ---------------------- input RMSNorm (+ flip copy for bwd) -----------------
__global__ __launch_bounds__(256)
void k_rmsnorm(const float* __restrict__ x, const float* __restrict__ w,
               u16* __restrict__ xnf, u16* __restrict__ xnb) {
    int bt = blockIdx.x;               // 0..NTOK-1
    int b = bt >> 11, t = bt & 2047;
    int tid = threadIdx.x;
    const float* row = x + (size_t)bt * DMODEL;
    float ss = 0.f;
    for (int i = tid; i < DMODEL; i += 256) { float v = row[i]; ss += v * v; }
    __shared__ float red[8];
    for (int off = 16; off > 0; off >>= 1) ss += __shfl_down(ss, off, 32);
    if ((tid & 31) == 0) red[tid >> 5] = ss;
    __syncthreads();
    if (tid == 0) { float s = 0; for (int i = 0; i < 8; ++i) s += red[i]; red[0] = s; }
    __syncthreads();
    float scale = rsqrtf(red[0] / (float)DMODEL + 1e-5f);
    size_t dF = (size_t)bt * DMODEL;
    size_t dB = ((size_t)b * SEQ + (SEQ - 1 - t)) * DMODEL;
    for (int i = tid; i < DMODEL; i += 256) {
        u16 v = f2bf(row[i] * scale * w[i]);
        xnf[dF + i] = v;
        xnb[dB + i] = v;
    }
}

// ---------------------- generic bf16 WMMA GEMM ------------------------------
// C[M,N](f32) = A[M,K](bf16) @ W[N,K](bf16)^T  (+bias[col]) (+resid[row,col])
// Double-buffered LDS, async global->LDS staging.
#define GBM 128
#define GBN 128
#define GBK 32
#define GLD 48      // LDS row stride (elems); 96B pitch -> 16B aligned rows
__global__ __launch_bounds__(256)
void k_gemm(const u16* __restrict__ A, const u16* __restrict__ W,
            float* __restrict__ C, const float* __restrict__ bias,
            const float* __restrict__ resid, int M, int N, int K) {
    __shared__ u16 sA[2][GBM * GLD];
    __shared__ u16 sW[2][GBN * GLD];
    int tid  = threadIdx.x;
    int lane = tid & 31;
    int w    = tid >> 5;           // wave 0..7
    int wm   = w >> 2;             // 0..1 : 64-row slice
    int wn   = w & 3;              // 0..3 : 32-col slice
    int bm   = blockIdx.x * GBM;
    int bn   = blockIdx.y * GBN;

    v8f acc[4][2];
    for (int i = 0; i < 4; ++i)
        for (int j = 0; j < 2; ++j) acc[i][j] = v8f_zero();

    // stage one 128x32 A tile and one 128x32 W tile into buffer `buf`
    auto stage = [&](int k0, int buf) {
        #pragma unroll
        for (int p = 0; p < 2; ++p) {
            int i = (p * 256 + tid) * 8;
            int r = i >> 5, c = i & 31;
            async_g2l_b128(&sA[buf][r * GLD + c],
                           A + (size_t)(bm + r) * K + k0 + c);
            int nr = bn + r;
            if (nr < N) {
                async_g2l_b128(&sW[buf][r * GLD + c],
                               W + (size_t)nr * K + k0 + c);
            } else {
                uint4 z = make_uint4(0u, 0u, 0u, 0u);
                *(uint4*)&sW[buf][r * GLD + c] = z;
            }
        }
    };

    int nk = K / GBK;
    stage(0, 0);
    wait_async0();
    __syncthreads();

    for (int kb = 0; kb < nk; ++kb) {
        int cur = kb & 1;
        if (kb + 1 < nk) stage((kb + 1) * GBK, cur ^ 1);   // overlap with WMMA

        v16bf af[4], bfm[2];
        #pragma unroll
        for (int i = 0; i < 4; ++i) af[i]  = ld_frag(sA[cur], wm * 64 + i * 16, 0, GLD);
        #pragma unroll
        for (int j = 0; j < 2; ++j) bfm[j] = ld_frag(sW[cur], wn * 32 + j * 16, 0, GLD);
        #pragma unroll
        for (int i = 0; i < 4; ++i)
            #pragma unroll
            for (int j = 0; j < 2; ++j)
                acc[i][j] = wmma_bf16(af[i], bfm[j], acc[i][j]);

        wait_async0();
        __syncthreads();
    }
    // store: VGPR ii -> M = ii + 8*(lane>=16); N = lane&15
    #pragma unroll
    for (int i = 0; i < 4; ++i)
        #pragma unroll
        for (int j = 0; j < 2; ++j) {
            int col = bn + wn * 32 + j * 16 + (lane & 15);
            if (col >= N) continue;
            #pragma unroll
            for (int ii = 0; ii < 8; ++ii) {
                int row = bm + wm * 64 + i * 16 + ii + ((lane >> 4) << 3);
                float v = acc[i][j][ii];
                if (bias)  v += bias[col];
                if (resid) v += resid[(size_t)row * N + col];
                C[(size_t)row * N + col] = v;
            }
        }
}

// ---------------------- conv1d(4) + SiLU + dt/softplus ----------------------
__global__ __launch_bounds__(256)
void k_conv(const float* __restrict__ zx, const float* __restrict__ cw,
            const float* __restrict__ cb, const float* __restrict__ dtb,
            const float* __restrict__ alog,
            float* __restrict__ adt, float* __restrict__ xs,
            u16* __restrict__ Xbf, u16* __restrict__ Bbf, u16* __restrict__ Cbf) {
    int bt = blockIdx.x;
    int b = bt >> 11, t = bt & 2047;
    int tid = threadIdx.x;
    __shared__ float sdt[NHEADS];
    const float* rowz = zx + (size_t)bt * NPROJ;
    if (tid < NHEADS) {
        float raw = rowz[2 * DINNER + 2 * DSTATE + tid] + dtb[tid];
        float d = (raw > 20.f) ? raw : log1pf(__expf(raw));   // softplus
        sdt[tid] = d;
        float A = -__expf(alog[tid]);
        adt[(size_t)bt * NHEADS + tid] = A * d;
    }
    __syncthreads();
    for (int ch = tid; ch < CONVCH; ch += 256) {
        float acc = cb[ch];
        #pragma unroll
        for (int k = 0; k < 4; ++k) {
            int tt = t + k - 3;
            float xv = (tt >= 0)
                ? zx[((size_t)b * SEQ + tt) * NPROJ + DINNER + ch] : 0.f;
            acc += xv * cw[ch * 4 + k];
        }
        float s = siluf(acc);
        if (ch < DINNER) {
            xs[(size_t)bt * DINNER + ch] = s;
            Xbf[(size_t)bt * DINNER + ch] = f2bf(s * sdt[ch >> 6]);
        } else if (ch < DINNER + DSTATE) {
            Bbf[(size_t)bt * DSTATE + (ch - DINNER)] = f2bf(s);
        } else {
            Cbf[(size_t)bt * DSTATE + (ch - DINNER - DSTATE)] = f2bf(s);
        }
    }
}

// ---------------------- per-chunk cumsum of A*dt ---------------------------
__global__ void k_cumsum(const float* __restrict__ adt,
                         float* __restrict__ acs, float* __restrict__ cdec) {
    int bid = blockIdx.x;              // (b*NH + h)*NC + c
    int c = bid & 15, h = (bid >> 4) & 31, b = bid >> 9;
    int i = threadIdx.x;               // 0..127
    __shared__ float s[CHUNK];
    s[i] = adt[((size_t)(b * SEQ + c * CHUNK + i)) * NHEADS + h];
    __syncthreads();
    for (int off = 1; off < CHUNK; off <<= 1) {
        float tv = (i >= off) ? s[i - off] : 0.f;
        __syncthreads();
        s[i] += tv;
        __syncthreads();
    }
    acs[(size_t)bid * CHUNK + i] = s[i];
    if (i == CHUNK - 1) cdec[bid] = __expf(s[i]);
}

// ---------------------- SSD diagonal block + chunk states ------------------
// One block per (b, chunk, head).  Three WMMA phases on LDS-resident tiles:
//   G = Cc @ Bc^T            (128x128x128)
//   Yd = (L o G) @ X         (128x64x128)   + D*xs skip
//   states = (diag(decay)X)^T @ Bc  (64x128x128)
#define TLD 136     // 272B pitch, 16B aligned
#define SSD_DIAG_LDS ((4*128 + 2*64) * TLD * 2 + CHUNK * 4)
__global__ __launch_bounds__(256)
void k_ssd_diag(const u16* __restrict__ Cb, const u16* __restrict__ Bb,
                const u16* __restrict__ Xb, const float* __restrict__ acs,
                const float* __restrict__ xs, const float* __restrict__ Dp,
                float* __restrict__ Y, float* __restrict__ states) {
    extern __shared__ char smem[];
    u16* sC   = (u16*)smem;                 // [128][TLD]  Cc[l][n]
    u16* sB   = sC  + 128 * TLD;            // [128][TLD]  Bc[s][n]
    u16* sBt  = sB  + 128 * TLD;            // [128][TLD]  Bc^T[n][l]
    u16* sG   = sBt + 128 * TLD;            // [128][TLD]  masked scores (bf16)
    u16* sXt  = sG  + 128 * TLD;            // [64][TLD]   X^T[p][s]
    u16* sXtd = sXt + 64 * TLD;             // [64][TLD]   decay-scaled X^T
    float* sA = (float*)(sXtd + 64 * TLD);  // [128] Acs

    int blk = blockIdx.x;                   // (b*NC + c)*NH + h
    int h = blk & 31, c = (blk >> 5) & 15, b = blk >> 9;
    int tid = threadIdx.x, lane = tid & 31, w = tid >> 5;

    const float* acsRow = acs + ((size_t)(b * NHEADS + h) * NCHUNK + c) * CHUNK;
    size_t rowBase = (size_t)b * SEQ + c * CHUNK;

    // async-stage Cc tile (no transform needed -> direct-to-LDS path)
    for (int i = tid * 8; i < 128 * 128; i += 256 * 8) {
        int r = i >> 7, col = i & 127;
        async_g2l_b128(&sC[r * TLD + col], Cb + (rowBase + r) * DSTATE + col);
    }

    if (tid < CHUNK) sA[tid] = acsRow[tid];
    __syncthreads();
    float acsLast = sA[CHUNK - 1];

    // stage B (and B^T) -- values needed in VGPRs for transpose
    for (int i = tid * 8; i < 128 * 128; i += 256 * 8) {
        int r = i >> 7, col = i & 127;
        uint4 bv = *(const uint4*)(Bb + (rowBase + r) * DSTATE + col);
        *(uint4*)&sB[r * TLD + col] = bv;
        union { uint4 q; u16 e[8]; } tv; tv.q = bv;
        #pragma unroll
        for (int j = 0; j < 8; ++j) sBt[(col + j) * TLD + r] = tv.e[j];
    }
    // stage X^T and decay-scaled X^T
    for (int i = tid * 8; i < 128 * 64; i += 256 * 8) {
        int s = i >> 6, p = i & 63;
        uint4 xv = *(const uint4*)(Xb + (rowBase + s) * DINNER + h * HEADDIM + p);
        union { uint4 q; u16 e[8]; } tv; tv.q = xv;
        float decay = __expf(acsLast - sA[s]);
        #pragma unroll
        for (int j = 0; j < 8; ++j) {
            sXt [(p + j) * TLD + s] = tv.e[j];
            sXtd[(p + j) * TLD + s] = f2bf(bf2f(tv.e[j]) * decay);
        }
    }
    wait_async0();
    __syncthreads();

    // ---- phase 1: G = Cc @ Bc^T -------------------------------------------
    int l0 = w * 16;
    v8f acc[8];
    #pragma unroll
    for (int nt = 0; nt < 8; ++nt) acc[nt] = v8f_zero();
    for (int k0 = 0; k0 < 128; k0 += 32) {
        v16bf a = ld_frag(sC, l0, k0, TLD);
        #pragma unroll
        for (int nt = 0; nt < 8; ++nt) {
            v16bf bb = ld_frag(sB, nt * 16, k0, TLD);
            acc[nt] = wmma_bf16(a, bb, acc[nt]);
        }
    }
    // causal decay mask, store bf16 scores
    int lBase = l0 + ((lane >> 4) << 3);
    #pragma unroll
    for (int nt = 0; nt < 8; ++nt) {
        int s = nt * 16 + (lane & 15);
        float as = sA[s];
        #pragma unroll
        for (int ii = 0; ii < 8; ++ii) {
            int l = lBase + ii;
            float g = (l >= s) ? acc[nt][ii] * __expf(sA[l] - as) : 0.f;
            sG[l * TLD + s] = f2bf(g);
        }
    }
    __syncthreads();

    // ---- phase 2: Yd = G @ X  (B operand = X^T rows p) --------------------
    v8f acc2[4];
    #pragma unroll
    for (int j = 0; j < 4; ++j) acc2[j] = v8f_zero();
    for (int k0 = 0; k0 < 128; k0 += 32) {
        v16bf a = ld_frag(sG, l0, k0, TLD);
        #pragma unroll
        for (int j = 0; j < 4; ++j) {
            v16bf bb = ld_frag(sXt, j * 16, k0, TLD);
            acc2[j] = wmma_bf16(a, bb, acc2[j]);
        }
    }
    float Dv = Dp[h];
    #pragma unroll
    for (int j = 0; j < 4; ++j) {
        int p = j * 16 + (lane & 15);
        #pragma unroll
        for (int ii = 0; ii < 8; ++ii) {
            int l = lBase + ii;
            size_t idx = (rowBase + l) * DINNER + h * HEADDIM + p;
            Y[idx] = acc2[j][ii] + Dv * xs[idx];
        }
    }

    // ---- phase 3: states = Xtd @ Bt ---------------------------------------
    v8f acc3[4];
    #pragma unroll
    for (int j = 0; j < 4; ++j) acc3[j] = v8f_zero();
    int p0 = (w & 3) * 16;
    int ntB = (w >> 2) * 4;
    for (int k0 = 0; k0 < 128; k0 += 32) {
        v16bf a = ld_frag(sXtd, p0, k0, TLD);
        #pragma unroll
        for (int j = 0; j < 4; ++j) {
            v16bf bb = ld_frag(sBt, (ntB + j) * 16, k0, TLD);
            acc3[j] = wmma_bf16(a, bb, acc3[j]);
        }
    }
    float* stBase = states + ((((size_t)b * NCHUNK + c) * NHEADS + h) * HEADDIM) * DSTATE;
    #pragma unroll
    for (int j = 0; j < 4; ++j) {
        int n = (ntB + j) * 16 + (lane & 15);
        #pragma unroll
        for (int ii = 0; ii < 8; ++ii) {
            int p = p0 + ii + ((lane >> 4) << 3);
            stBase[(size_t)p * DSTATE + n] = acc3[j][ii];
        }
    }
}

// ---------------------- sequential inter-chunk scan (in place) --------------
__global__ __launch_bounds__(256)
void k_scan(float* __restrict__ states, const float* __restrict__ cdec) {
    int bh = blockIdx.x;               // b*NH + h
    int b = bh >> 5, h = bh & 31;
    int tid = threadIdx.x;
    float carry[32];
    #pragma unroll
    for (int e = 0; e < 32; ++e) carry[e] = 0.f;
    for (int c = 0; c < NCHUNK; ++c) {
        float cd = cdec[(size_t)bh * NCHUNK + c];
        float* base = states + ((((size_t)b * NCHUNK + c) * NHEADS + h) * HEADDIM) * DSTATE;
        #pragma unroll
        for (int e = 0; e < 32; ++e) {
            int idx = e * 256 + tid;   // covers 64*128 = 8192
            float st = base[idx];
            base[idx] = carry[e];      // prev-state for this chunk
            carry[e] = carry[e] * cd + st;
        }
    }
}

// ---------------------- SSD off-diagonal: Y += (C*exp(Acs)) @ prev^T --------
#define SSD_OFF_LDS ((128 + 64) * TLD * 2 + CHUNK * 4)
__global__ __launch_bounds__(256)
void k_ssd_off(const u16* __restrict__ Cb, const float* __restrict__ acs,
               const float* __restrict__ prev, float* __restrict__ Y) {
    extern __shared__ char smem[];
    u16* sCe = (u16*)smem;                  // [128][TLD] exp(Acs[l])*Cc[l][n]
    u16* sP  = sCe + 128 * TLD;             // [64][TLD]  prev[p][n]
    float* sA = (float*)(sP + 64 * TLD);

    int blk = blockIdx.x;
    int h = blk & 31, c = (blk >> 5) & 15, b = blk >> 9;
    int tid = threadIdx.x, lane = tid & 31, w = tid >> 5;

    const float* acsRow = acs + ((size_t)(b * NHEADS + h) * NCHUNK + c) * CHUNK;
    if (tid < CHUNK) sA[tid] = acsRow[tid];
    __syncthreads();

    size_t rowBase = (size_t)b * SEQ + c * CHUNK;
    for (int i = tid * 8; i < 128 * 128; i += 256 * 8) {
        int r = i >> 7, col = i & 127;
        uint4 cv = *(const uint4*)(Cb + (rowBase + r) * DSTATE + col);
        union { uint4 q; u16 e[8]; } tv; tv.q = cv;
        float e = __expf(sA[r]);
        #pragma unroll
        for (int j = 0; j < 8; ++j)
            sCe[r * TLD + col + j] = f2bf(bf2f(tv.e[j]) * e);
    }
    const float* pv = prev + ((((size_t)b * NCHUNK + c) * NHEADS + h) * HEADDIM) * DSTATE;
    for (int i = tid * 4; i < 64 * 128; i += 256 * 4) {
        int p = i >> 7, n = i & 127;
        float4 v4 = *(const float4*)(pv + (size_t)p * DSTATE + n);
        sP[p * TLD + n + 0] = f2bf(v4.x);
        sP[p * TLD + n + 1] = f2bf(v4.y);
        sP[p * TLD + n + 2] = f2bf(v4.z);
        sP[p * TLD + n + 3] = f2bf(v4.w);
    }
    __syncthreads();

    int l0 = w * 16;
    v8f acc[4];
    #pragma unroll
    for (int j = 0; j < 4; ++j) acc[j] = v8f_zero();
    for (int k0 = 0; k0 < 128; k0 += 32) {
        v16bf a = ld_frag(sCe, l0, k0, TLD);
        #pragma unroll
        for (int j = 0; j < 4; ++j) {
            v16bf bb = ld_frag(sP, j * 16, k0, TLD);
            acc[j] = wmma_bf16(a, bb, acc[j]);
        }
    }
    int lBase = l0 + ((lane >> 4) << 3);
    #pragma unroll
    for (int j = 0; j < 4; ++j) {
        int p = j * 16 + (lane & 15);
        #pragma unroll
        for (int ii = 0; ii < 8; ++ii) {
            int l = lBase + ii;
            size_t idx = (rowBase + l) * DINNER + h * HEADDIM + p;
            Y[idx] += acc[j][ii];
        }
    }
}

// ---------------------- gating + RMSNorm -> bf16 ----------------------------
__global__ __launch_bounds__(256)
void k_gate_norm(const float* __restrict__ Y, const float* __restrict__ zx,
                 const float* __restrict__ gw, u16* __restrict__ yg) {
    int bt = blockIdx.x;
    int tid = threadIdx.x;
    const float* yrow = Y + (size_t)bt * DINNER;
    const float* zrow = zx + (size_t)bt * NPROJ;   // z = first DINNER cols
    float ss = 0.f;
    for (int ch = tid; ch < DINNER; ch += 256) {
        float v = yrow[ch] * siluf(zrow[ch]);
        ss += v * v;
    }
    __shared__ float red[8];
    for (int off = 16; off > 0; off >>= 1) ss += __shfl_down(ss, off, 32);
    if ((tid & 31) == 0) red[tid >> 5] = ss;
    __syncthreads();
    if (tid == 0) { float s = 0; for (int i = 0; i < 8; ++i) s += red[i]; red[0] = s; }
    __syncthreads();
    float scale = rsqrtf(red[0] / (float)DINNER + 1e-5f);
    for (int ch = tid; ch < DINNER; ch += 256) {
        float v = yrow[ch] * siluf(zrow[ch]);
        yg[(size_t)bt * DINNER + ch] = f2bf(v * scale * gw[ch]);
    }
}

// ---------------------- concat (with bwd flip) into bf16 --------------------
__global__ void k_concat(const float* __restrict__ ydir, u16* __restrict__ ycat,
                         int dir) {
    int i = blockIdx.x * 256 + threadIdx.x;
    if (i >= NTOK * DMODEL) return;
    int bt = i >> 10, col = i & 1023;
    int b = bt >> 11, t = bt & 2047;
    int tdst = dir ? (SEQ - 1 - t) : t;
    ycat[((size_t)(b * SEQ + tdst)) * (2 * DMODEL) + dir * DMODEL + col] =
        f2bf(ydir[i]);
}

// ---------------------------------------------------------------------------
extern "C" void kernel_launch(void* const* d_in, const int* in_sizes, int n_in,
                              void* d_out, int out_size, void* d_ws, size_t ws_size,
                              hipStream_t stream) {
    (void)in_sizes; (void)n_in; (void)out_size; (void)ws_size;

    const float* x        = (const float*)d_in[0];
    const float* norm_w   = (const float*)d_in[1];
    const float* ow_blk   = (const float*)d_in[2];
    const float* ob_blk   = (const float*)d_in[3];

    // workspace carve-up (256B aligned)
    char* wp = (char*)d_ws;
    auto alloc = [&](size_t bytes) -> char* {
        char* p = wp; wp += (bytes + 255) & ~(size_t)255; return p;
    };
    u16*   wbf_in[2]; wbf_in[0] = (u16*)alloc((size_t)NPROJ * DMODEL * 2);
                      wbf_in[1] = (u16*)alloc((size_t)NPROJ * DMODEL * 2);
    u16*   wbf_out[2]; wbf_out[0] = (u16*)alloc((size_t)DMODEL * DINNER * 2);
                       wbf_out[1] = (u16*)alloc((size_t)DMODEL * DINNER * 2);
    u16*   wbf_blk  = (u16*)alloc((size_t)DMODEL * 2 * DMODEL * 2);
    u16*   xn[2];  xn[0] = (u16*)alloc((size_t)NTOK * DMODEL * 2);
                   xn[1] = (u16*)alloc((size_t)NTOK * DMODEL * 2);
    float* zxb  = (float*)alloc((size_t)NTOK * NPROJ * 4);
    float* adt  = (float*)alloc((size_t)NTOK * NHEADS * 4);
    float* acs  = (float*)alloc((size_t)BATCH * NHEADS * NCHUNK * CHUNK * 4);
    float* cdec = (float*)alloc((size_t)BATCH * NHEADS * NCHUNK * 4);
    float* xsF  = (float*)alloc((size_t)NTOK * DINNER * 4);
    u16*   Xbf  = (u16*)alloc((size_t)NTOK * DINNER * 2);
    u16*   Bbf  = (u16*)alloc((size_t)NTOK * DSTATE * 2);
    u16*   Cbf  = (u16*)alloc((size_t)NTOK * DSTATE * 2);
    float* Yb   = (float*)alloc((size_t)NTOK * DINNER * 4);
    float* stt  = (float*)alloc((size_t)BATCH * NCHUNK * NHEADS * HEADDIM * DSTATE * 4);
    u16*   yg   = (u16*)alloc((size_t)NTOK * DINNER * 2);
    float* ydir = (float*)alloc((size_t)NTOK * DMODEL * 4);
    u16*   ycat = (u16*)alloc((size_t)NTOK * 2 * DMODEL * 2);

    // allow >64KB dynamic LDS on the SSD kernels
    hipFuncSetAttribute((const void*)k_ssd_diag,
                        hipFuncAttributeMaxDynamicSharedMemorySize, SSD_DIAG_LDS);
    hipFuncSetAttribute((const void*)k_ssd_off,
                        hipFuncAttributeMaxDynamicSharedMemorySize, SSD_OFF_LDS);

    auto cvt = [&](const float* s, u16* d, int n) {
        k_f32_to_bf16<<<(n + 255) / 256, 256, 0, stream>>>(s, d, n);
    };
    cvt((const float*)d_in[4],  wbf_in[0],  NPROJ * DMODEL);
    cvt((const float*)d_in[12], wbf_in[1],  NPROJ * DMODEL);
    cvt((const float*)d_in[11], wbf_out[0], DMODEL * DINNER);
    cvt((const float*)d_in[19], wbf_out[1], DMODEL * DINNER);
    cvt(ow_blk, wbf_blk, DMODEL * 2 * DMODEL);

    k_rmsnorm<<<NTOK, 256, 0, stream>>>(x, norm_w, xn[0], xn[1]);

    for (int dir = 0; dir < 2; ++dir) {
        int o = 4 + dir * 8;
        const float* conv_w  = (const float*)d_in[o + 1];
        const float* conv_b  = (const float*)d_in[o + 2];
        const float* dt_bias = (const float*)d_in[o + 3];
        const float* A_log   = (const float*)d_in[o + 4];
        const float* Dvec    = (const float*)d_in[o + 5];
        const float* gnorm_w = (const float*)d_in[o + 6];

        // in_proj: [4096,1024] @ [4384,1024]^T
        k_gemm<<<dim3(NTOK / GBM, (NPROJ + GBN - 1) / GBN), 256, 0, stream>>>(
            xn[dir], wbf_in[dir], zxb, nullptr, nullptr, NTOK, NPROJ, DMODEL);

        k_conv<<<NTOK, 256, 0, stream>>>(zxb, conv_w, conv_b, dt_bias, A_log,
                                         adt, xsF, Xbf, Bbf, Cbf);

        k_cumsum<<<BATCH * NHEADS * NCHUNK, CHUNK, 0, stream>>>(adt, acs, cdec);

        k_ssd_diag<<<BATCH * NCHUNK * NHEADS, 256, SSD_DIAG_LDS, stream>>>(
            Cbf, Bbf, Xbf, acs, xsF, Dvec, Yb, stt);

        k_scan<<<BATCH * NHEADS, 256, 0, stream>>>(stt, cdec);

        k_ssd_off<<<BATCH * NCHUNK * NHEADS, 256, SSD_OFF_LDS, stream>>>(
            Cbf, acs, stt, Yb);

        k_gate_norm<<<NTOK, 256, 0, stream>>>(Yb, zxb, gnorm_w, yg);

        // out_proj: [4096,2048] @ [1024,2048]^T
        k_gemm<<<dim3(NTOK / GBM, DMODEL / GBN), 256, 0, stream>>>(
            yg, wbf_out[dir], ydir, nullptr, nullptr, NTOK, DMODEL, DINNER);

        k_concat<<<(NTOK * DMODEL + 255) / 256, 256, 0, stream>>>(ydir, ycat, dir);
    }

    // final: x + cat @ out_w_blk^T + out_b_blk
    k_gemm<<<dim3(NTOK / GBM, DMODEL / GBN), 256, 0, stream>>>(
        ycat, wbf_blk, (float*)d_out, ob_blk, x, NTOK, DMODEL, 2 * DMODEL);
}
